// kqvAttn_25151328485711
// MI455X (gfx1250) — matrix operand order repaired and checked by reading, not compiled, not run on hardware
//
#include <hip/hip_runtime.h>
#include <hip/hip_bf16.h>
#include <math.h>
#include <stdint.h>

// Problem sizes (compile-time, match reference)
#define G     8192
#define IND   512
#define KQD   128
#define OUTD  512
#define EPSF  1e-12f

typedef __attribute__((ext_vector_type(16))) __bf16 bf16x16;
typedef __attribute__((ext_vector_type(8)))  float  f32x8;

// ---------------------------------------------------------------------------
// WMMA fragment loaders for V_WMMA_F32_16X16X32_BF16 (wave32).
// A as loaded here (16x32, M x K): lane m = M (m = lane&15), element e -> K =
//   e + 16*half   (half = lane>=16): contiguous k order, not AMD's table order
// B as loaded here (32x16, K x N): lane n = N, element e -> K = e + 16*half (same k order as A)
// C/D layout (16x16 f32): VGPR v, lane -> M = v + 8*half, N = lane&15
// ---------------------------------------------------------------------------

__device__ __forceinline__ bf16x16
load_a_bf(const __bf16* base, int ld, int row, int kbase, int hf) {
  const __bf16* p = base + (size_t)row * ld + kbase + 16 * hf;
  bf16x16 a;
#pragma unroll
  for (int e = 0; e < 16; ++e) a[e] = p[e];
  return a;
}

__device__ __forceinline__ bf16x16
load_b_bf(const __bf16* base, int ld, int row, int kbase, int hf) {
  const __bf16* p = base + (size_t)row * ld + kbase + 16 * hf;
  bf16x16 b;
#pragma unroll
  for (int e = 0; e < 16; ++e) b[e] = p[e];
  return b;
}

__device__ __forceinline__ bf16x16
load_a_f32(const float* base, int ld, int row, int kbase, int hf) {
  const float* p = base + (size_t)row * ld + kbase + 16 * hf;
  bf16x16 a;
#pragma unroll
  for (int e = 0; e < 16; ++e) a[e] = (__bf16)p[e];
  return a;
}

__device__ __forceinline__ bf16x16
load_b_f32(const float* base, int ld, int row, int kbase, int hf) {
  const float* p = base + (size_t)row * ld + kbase + 16 * hf;
  bf16x16 b;
#pragma unroll
  for (int e = 0; e < 16; ++e) b[e] = (__bf16)p[e];
  return b;
}

#define WMMA_BF16(a, b, c) \
  __builtin_amdgcn_wmma_f32_16x16x32_bf16(false, (a), false, (b), (short)0, (c), false, false)

// --- gfx1250 async copy: 16 bytes/lane, global -> LDS, tracked by ASYNCcnt ---
__device__ __forceinline__ void async_g2l_b128(uint32_t lds_byte_addr, const void* gaddr) {
  asm volatile("global_load_async_to_lds_b128 %0, %1, off"
               :: "v"(lds_byte_addr), "v"(gaddr)
               : "memory");
}
__device__ __forceinline__ void wait_asynccnt0() {
  asm volatile("s_wait_asynccnt 0x0" ::: "memory");
}

// ---------------------------------------------------------------------------
// Pass A: K = xWk^T+bk, Q = xWq^T+bq, V = xWv^T+bv.
// Writes bf16 K [G][128], bf16 Q [G][128], bf16 V^T [512][G], f32 V [G][512].
// Grid: (12 col-blocks of 64 over concat cols 0..767, 128 row-blocks of 64).
// Block: 128 threads = 4 waves; wave = row-tile, 4 col-tiles each.
// ---------------------------------------------------------------------------
__global__ void __launch_bounds__(128)
kqv_proj_kernel(const float* __restrict__ x,
                const float* __restrict__ Wk, const float* __restrict__ bk,
                const float* __restrict__ Wq, const float* __restrict__ bq,
                const float* __restrict__ Wv, const float* __restrict__ bv,
                __bf16* __restrict__ Kbf, __bf16* __restrict__ Qbf,
                __bf16* __restrict__ Vt,  float* __restrict__ Vf) {
  const int lane = threadIdx.x & 31;
  const int wave = threadIdx.x >> 5;
  const int hf = lane >> 4;          // half of wave
  const int hl = lane & 15;          // lane within half
  const int rb = blockIdx.y;         // row block (64 rows)
  const int cb = blockIdx.x;         // col block (64 of 768 concat cols)
  const int row0 = rb * 64 + wave * 16;

  f32x8 acc[4] = {};
  for (int kb = 0; kb < IND; kb += 32) {
    bf16x16 a = load_a_f32(x, IND, row0 + hl, kb, hf);
#pragma unroll
    for (int ct = 0; ct < 4; ++ct) {
      const int gc0 = cb * 64 + ct * 16;
      const float* W;
      int o0;
      if (gc0 < 128)      { W = Wk; o0 = gc0; }
      else if (gc0 < 256) { W = Wq; o0 = gc0 - 128; }
      else                { W = Wv; o0 = gc0 - 256; }
      bf16x16 b = load_b_f32(W, IND, o0 + hl, kb, hf);
      acc[ct] = WMMA_BF16(a, b, acc[ct]);
    }
  }
#pragma unroll
  for (int ct = 0; ct < 4; ++ct) {
    const int gc = cb * 64 + ct * 16 + hl;   // this lane's concat column
#pragma unroll
    for (int v = 0; v < 8; ++v) {
      const int row = row0 + v + 8 * hf;
      if (gc < 128) {
        Kbf[(size_t)row * KQD + gc] = (__bf16)(acc[ct][v] + bk[gc]);
      } else if (gc < 256) {
        const int o = gc - 128;
        Qbf[(size_t)row * KQD + o] = (__bf16)(acc[ct][v] + bq[o]);
      } else {
        const int o = gc - 256;
        const float val = acc[ct][v] + bv[o];
        Vt[(size_t)o * G + row] = (__bf16)val;   // transposed bf16 for WMMA-B
        Vf[(size_t)row * OUTD + o] = val;        // f32 for residual
      }
    }
  }
}

// ---------------------------------------------------------------------------
// Pass B: norm2[i] = sum_j (K_i . Q_j)^2 over all 8192 j.
// Grid: 128 row-blocks of 64. Block: 256 threads = 8 waves.
// Wave w: row-tile w&3, col-tile group w>>2 (4 of 8 j-tiles per 128-chunk).
// Q stays in L2 (2 MB); direct global B-fragment loads.
// ---------------------------------------------------------------------------
__global__ void __launch_bounds__(256)
kq_norm_kernel(const __bf16* __restrict__ Kbf, const __bf16* __restrict__ Qbf,
               float* __restrict__ norm2) {
  const int lane = threadIdx.x & 31;
  const int wave = threadIdx.x >> 5;
  const int hf = lane >> 4, hl = lane & 15;
  const int rb = blockIdx.x;
  const int rt = wave & 3, cg = wave >> 2;
  const int rowA = rb * 64 + rt * 16;

  bf16x16 kfr[4];
#pragma unroll
  for (int kb = 0; kb < 4; ++kb) kfr[kb] = load_a_bf(Kbf, KQD, rowA + hl, kb * 32, hf);

  float ss[8] = {0.f, 0.f, 0.f, 0.f, 0.f, 0.f, 0.f, 0.f};
  for (int j0 = 0; j0 < G; j0 += 128) {
#pragma unroll
    for (int t = 0; t < 4; ++t) {
      const int n0 = j0 + (cg * 4 + t) * 16;
      f32x8 c = {};
#pragma unroll
      for (int kb = 0; kb < 4; ++kb) {
        bf16x16 b = load_b_bf(Qbf, KQD, n0 + hl, kb * 32, hf);
        c = WMMA_BF16(kfr[kb], b, c);
      }
#pragma unroll
      for (int v = 0; v < 8; ++v) ss[v] += c[v] * c[v];
    }
  }
  // reduce over the 16 lanes of each half (columns), then atomically add rows
#pragma unroll
  for (int v = 0; v < 8; ++v) {
    ss[v] += __shfl_xor(ss[v], 1, 32);
    ss[v] += __shfl_xor(ss[v], 2, 32);
    ss[v] += __shfl_xor(ss[v], 4, 32);
    ss[v] += __shfl_xor(ss[v], 8, 32);
  }
  if (hl == 0) {
#pragma unroll
    for (int v = 0; v < 8; ++v)
      atomicAdd(&norm2[rowA + v + 8 * hf], ss[v]);
  }
}

// ---------------------------------------------------------------------------
// Pass C: out = (K Q^T / max(norm,eps)) V + V, fused flash-style.
// Grid: 128 row-blocks of 64. Block: 512 threads = 16 waves.
// Per 128-wide j chunk:
//   1) S tiles from LDS-staged Q (async-copied, bank-conflict-free 136 stride)
//   2) S (bf16) -> LDS, barrier
//   3) issue async copy of NEXT Q chunk  (overlaps with step 4)
//   4) S @ V WMMAs against L2-resident V^T
//   5) s_wait_asynccnt 0 happens at top of next iteration, before the barrier
// ---------------------------------------------------------------------------
#define QS_LD 136   // padded bf16 row stride: 272B = 68 dwords -> conflict-free
__global__ void __launch_bounds__(512)
attn_out_kernel(const __bf16* __restrict__ Kbf, const __bf16* __restrict__ Qbf,
                const __bf16* __restrict__ Vt, const float* __restrict__ Vf,
                const float* __restrict__ norm2, float* __restrict__ out) {
  __shared__ __bf16 Qs[128][QS_LD];  // 34 KB staged Q chunk
  __shared__ __bf16 Sld[64][136];    // 17 KB S tile

  const int lane = threadIdx.x & 31;
  const int wave = threadIdx.x >> 5;
  const int hf = lane >> 4, hl = lane & 15;
  const int rb = blockIdx.x;
  const int rt = wave & 3;          // row tile (shared with A-frags)
  const int sg = wave >> 2;         // 0..3: column / j-tile group
  const int rowA = rb * 64 + rt * 16;

  const uint32_t qs_base = (uint32_t)(uintptr_t)&Qs[0][0];
  // this lane's slice of the cooperative row-wise async copy:
  // per wave: 4 instructions, each moving 2 rows x 256B (16B per lane)
  const int cp_half = lane >> 4;           // which of the 2 rows
  const int cp_byte = (lane & 15) * 16;    // 16B slot within the 256B row

  // prefetch chunk 0
#pragma unroll
  for (int q = 0; q < 4; ++q) {
    const int r = wave * 8 + q * 2 + cp_half;
    async_g2l_b128(qs_base + (uint32_t)(r * (QS_LD * 2) + cp_byte),
                   (const char*)Qbf + (size_t)r * (KQD * 2) + cp_byte);
  }

  bf16x16 kfr[4];
#pragma unroll
  for (int kb = 0; kb < 4; ++kb) kfr[kb] = load_a_bf(Kbf, KQD, rowA + hl, kb * 32, hf);

  f32x8 acc[8] = {};

  for (int j0 = 0; j0 < G; j0 += 128) {
    wait_asynccnt0();     // my slice of this chunk has landed in LDS
    __syncthreads();      // everyone's slice landed; Sld from prev iter free

    // --- S tile: this wave computes (rt, jt) for jt = sg*2, sg*2+1 ---
#pragma unroll
    for (int u = 0; u < 2; ++u) {
      const int jt = sg * 2 + u;
      f32x8 c = {};
#pragma unroll
      for (int kb = 0; kb < 4; ++kb) {
        bf16x16 b = load_b_bf(&Qs[0][0], QS_LD, jt * 16 + hl, kb * 32, hf);
        c = WMMA_BF16(kfr[kb], b, c);
      }
#pragma unroll
      for (int v = 0; v < 8; ++v)
        Sld[rt * 16 + v + 8 * hf][jt * 16 + hl] = (__bf16)c[v];
    }
    __syncthreads();      // S ready; all waves done reading Qs

    // --- overlap: async-prefetch next Q chunk while we do S @ V ---
    if (j0 + 128 < G) {
#pragma unroll
      for (int q = 0; q < 4; ++q) {
        const int r = wave * 8 + q * 2 + cp_half;
        async_g2l_b128(qs_base + (uint32_t)(r * (QS_LD * 2) + cp_byte),
                       (const char*)Qbf + (size_t)(j0 + 128 + r) * (KQD * 2) + cp_byte);
      }
    }

    // --- S @ V: A = S rows (LDS), B = V^T rows (contiguous bf16, L2) ---
    bf16x16 sfr[4];
#pragma unroll
    for (int kb = 0; kb < 4; ++kb)
      sfr[kb] = load_a_bf(&Sld[0][0], 136, rt * 16 + hl, kb * 32, hf);
#pragma unroll
    for (int t = 0; t < 8; ++t) {
      const int ct = (t >> 1) * 8 + sg * 2 + (t & 1);  // wave's 8 col tiles
      const int c0 = ct * 16;
#pragma unroll
      for (int kb = 0; kb < 4; ++kb) {
        bf16x16 b = load_b_bf(Vt, G, c0 + hl, j0 + kb * 32, hf);
        acc[t] = WMMA_BF16(sfr[kb], b, acc[t]);
      }
    }
  }

  // epilogue: row scale + residual
  float inv[8];
#pragma unroll
  for (int v = 0; v < 8; ++v) {
    const float n = sqrtf(norm2[rowA + v + 8 * hf]);
    inv[v] = 1.0f / fmaxf(n, EPSF);
  }
#pragma unroll
  for (int t = 0; t < 8; ++t) {
    const int ct = (t >> 1) * 8 + sg * 2 + (t & 1);
    const int col = ct * 16 + hl;
#pragma unroll
    for (int v = 0; v < 8; ++v) {
      const int row = rowA + v + 8 * hf;
      out[(size_t)row * OUTD + col] =
          acc[t][v] * inv[v] + Vf[(size_t)row * OUTD + col];
    }
  }
}

// ---------------------------------------------------------------------------
// Workspace layout (bytes):
//   Kbf  @ 0        : G*128*2  =  2 MB  (bf16)
//   Qbf  @ 2 MB     : G*128*2  =  2 MB  (bf16)
//   Vt   @ 4 MB     : 512*G*2  =  8 MB  (bf16, transposed)
//   Vf   @ 12 MB    : G*512*4  = 16 MB  (f32)
//   norm2@ 28 MB    : G*4      = 32 KB  (f32)
// ---------------------------------------------------------------------------
extern "C" void kernel_launch(void* const* d_in, const int* in_sizes, int n_in,
                              void* d_out, int out_size, void* d_ws, size_t ws_size,
                              hipStream_t stream) {
  const float* x  = (const float*)d_in[0];
  const float* Wk = (const float*)d_in[1];
  const float* bk = (const float*)d_in[2];
  const float* Wq = (const float*)d_in[3];
  const float* bq = (const float*)d_in[4];
  const float* Wv = (const float*)d_in[5];
  const float* bv = (const float*)d_in[6];
  float* out = (float*)d_out;

  char* ws = (char*)d_ws;
  __bf16* Kbf  = (__bf16*)(ws);
  __bf16* Qbf  = (__bf16*)(ws + ((size_t)2 << 20));
  __bf16* Vt   = (__bf16*)(ws + ((size_t)4 << 20));
  float*  Vf   = (float*) (ws + ((size_t)12 << 20));
  float*  nrm2 = (float*) (ws + ((size_t)28 << 20));

  hipMemsetAsync(nrm2, 0, G * sizeof(float), stream);

  kqv_proj_kernel<<<dim3(12, 128), 128, 0, stream>>>(
      x, Wk, bk, Wq, bq, Wv, bv, Kbf, Qbf, Vt, Vf);

  kq_norm_kernel<<<dim3(128), 256, 0, stream>>>(Kbf, Qbf, nrm2);

  attn_out_kernel<<<dim3(128), 512, 0, stream>>>(Kbf, Qbf, Vt, Vf, nrm2, out);
}
